// AttentionBlock_29137058136076
// MI455X (gfx1250) — matrix operand order repaired and checked by reading, not compiled, not run on hardware
//
#include <hip/hip_runtime.h>
#include <hip/hip_bf16.h>

#define NB   8
#define SEQ  1024
#define DM   1024
#define NH   16
#define DH   64
#define NQKV 3072
#define RMS_EPS 1.1920929e-07f

typedef __attribute__((ext_vector_type(16))) __bf16 bf16x16;
typedef __attribute__((ext_vector_type(8)))  __bf16 bf16x8;
typedef __attribute__((ext_vector_type(8)))  float  f32x8;

// ---- WMMA fragment loaders (contiguous k order, same for A and B, wave32) ----
// A (16x32 bf16): lane holds row M=lane&15; elements 0..15 = K at k0..k0+15,
// where k0 = 16*(lane>>4).
__device__ __forceinline__ bf16x16 load_frag_a(const __bf16* base, int ld) {
  int lane = threadIdx.x & 31;
  int row  = lane & 15;
  int k0   = (lane >> 4) * 16;
  const __bf16* p = base + (size_t)row * ld + k0;
  bf16x8 lo = *(const bf16x8*)(p);
  bf16x8 hi = *(const bf16x8*)(p + 8);
  bf16x16 a;
#pragma unroll
  for (int i = 0; i < 8; ++i) { a[i] = lo[i]; a[i + 8] = hi[i]; }
  return a;
}

// B (32x16 bf16): lane holds column N=lane&15; 16 contiguous K values
// starting at K = 16*(lane>>4).  One 32-byte load.
__device__ __forceinline__ bf16x16 load_frag_b(const __bf16* base, int ld) {
  int lane = threadIdx.x & 31;
  int col  = lane & 15;
  int k0   = (lane >> 4) * 16;
  return *(const bf16x16*)(base + (size_t)col * ld + k0);
}

__device__ __forceinline__ f32x8 wmma_bf16(bf16x16 a, bf16x16 b, f32x8 c) {
  return __builtin_amdgcn_wmma_f32_16x16x32_bf16(false, a, false, b,
                                                 (short)0, c, false, false);
}

// ---- RMSNorm (fp32 in, bf16 out), one block per row ----
__global__ void rmsnorm_kernel(const float* __restrict__ x, __bf16* __restrict__ xn) {
  __shared__ float red[8];
  int row = blockIdx.x;
  const float* xr = x + (size_t)row * DM;
  float ss = 0.f;
  for (int i = threadIdx.x; i < DM; i += 256) { float v = xr[i]; ss += v * v; }
#pragma unroll
  for (int off = 16; off > 0; off >>= 1) ss += __shfl_xor(ss, off, 32);
  if ((threadIdx.x & 31) == 0) red[threadIdx.x >> 5] = ss;
  __syncthreads();
  float tot = 0.f;
#pragma unroll
  for (int w = 0; w < 8; ++w) tot += red[w];
  float scale = rsqrtf(tot * (1.0f / DM) + RMS_EPS);
  for (int i = threadIdx.x; i < DM; i += 256)
    xn[(size_t)row * DM + i] = (__bf16)(xr[i] * scale);
}

// ---- fp32 -> bf16 weight conversion ----
__global__ void f2bf_kernel(const float* __restrict__ in, __bf16* __restrict__ out, int n) {
  int i = blockIdx.x * blockDim.x + threadIdx.x;
  if (i < n) out[i] = (__bf16)in[i];
}

// ---- QKV GEMM: [8192x1024]bf16 @ Wqkv[3072x1024]^T, register-blocked 32Mx64N
// per wave (2x4 tiles, 8 accumulators).  De-interleave on store:
// n = h*192 + d*3 + j (j: 0=Q,1=K,2=V). V stored transposed [b,h,d,s].
__global__ void qkv_gemm_kernel(const __bf16* __restrict__ Xn,
                                const __bf16* __restrict__ W,
                                const float*  __restrict__ bqkv,
                                __bf16* __restrict__ Q, __bf16* __restrict__ K,
                                __bf16* __restrict__ Vt) {
  const int NBLK = NQKV / 64;                // 48 n-blocks of 64
  int t  = blockIdx.x * 8 + (threadIdx.x >> 5);
  int mb = t / NBLK, nb = t % NBLK;
  const __bf16* arow = Xn + (size_t)mb * 32 * DM;
  const __bf16* brow = W  + (size_t)nb * 64 * DM;
  f32x8 acc[2][4] = {};
  for (int k = 0; k < DM; k += 32) {
    if (k + 64 < DM) {
      __builtin_prefetch(arow + k + 64, 0, 1);
      __builtin_prefetch(brow + k + 64, 0, 1);
    }
    bf16x16 a0 = load_frag_a(arow + k, DM);
    bf16x16 a1 = load_frag_a(arow + 16 * DM + k, DM);
#pragma unroll
    for (int j = 0; j < 4; ++j) {
      bf16x16 b = load_frag_b(brow + (size_t)j * 16 * DM + k, DM);
      acc[0][j] = wmma_bf16(a0, b, acc[0][j]);
      acc[1][j] = wmma_bf16(a1, b, acc[1][j]);
    }
  }
  int lane = threadIdx.x & 31;
  int hi = lane >> 4;
#pragma unroll
  for (int j = 0; j < 4; ++j) {
    int n = nb * 64 + j * 16 + (lane & 15);
    int h = n / 192, rem = n % 192, d = rem / 3, jj = rem % 3;
    float bias = bqkv[n];
#pragma unroll
    for (int i = 0; i < 2; ++i)
#pragma unroll
      for (int r = 0; r < 8; ++r) {
        int m  = mb * 32 + i * 16 + r + hi * 8;   // global row (b*S + s)
        int b_ = m >> 10, s = m & 1023;
        float v = acc[i][j][r] + bias;
        size_t bh = (size_t)b_ * NH + h;
        if      (jj == 0) Q [(bh * SEQ + s) * DH + d] = (__bf16)v;
        else if (jj == 1) K [(bh * SEQ + s) * DH + d] = (__bf16)v;
        else              Vt[(bh * DH  + d) * SEQ + s] = (__bf16)v;  // transposed
      }
  }
}

// ---- Attention: one workgroup (8 waves) per (b,h).
// Column-axis softmax (softmax over q for each key column, causal q>=k).
__global__ void attention_kernel(const __bf16* __restrict__ Q,
                                 const __bf16* __restrict__ K,
                                 const __bf16* __restrict__ Vt,
                                 __bf16* __restrict__ Z) {
  __shared__ float  sm[SEQ];                 // per-column max
  __shared__ float  sl[SEQ];                 // per-column 1/denom
  __shared__ __bf16 ptile[8][16 * 32];       // per-wave P staging (D->A layout)

  int bh = blockIdx.x;
  const __bf16* Qh = Q  + (size_t)bh * SEQ * DH;
  const __bf16* Kh = K  + (size_t)bh * SEQ * DH;
  const __bf16* Vh = Vt + (size_t)bh * DH * SEQ;
  int b_ = bh >> 4, h = bh & 15;
  int wv   = threadIdx.x >> 5;
  int lane = threadIdx.x & 31;
  int n = lane & 15, hi = lane >> 4;
  const int NT = SEQ / 16;                   // 64 tiles
  const float NEG_INF = -__builtin_inff();

  // masked 16x16 score tile (2 WMMAs over d)
  auto score_tile = [&](int qt, int kt) -> f32x8 {
    f32x8 acc = {};
#pragma unroll
    for (int dk = 0; dk < DH; dk += 32) {
      bf16x16 a = load_frag_a(Qh + (size_t)qt * 16 * DH + dk, DH);
      bf16x16 b = load_frag_b(Kh + (size_t)kt * 16 * DH + dk, DH);
      acc = wmma_bf16(a, b, acc);
    }
    int kg = kt * 16 + n;
#pragma unroll
    for (int r = 0; r < 8; ++r) {
      int qg = qt * 16 + r + hi * 8;
      if (kg > qg) acc[r] = NEG_INF;
    }
    return acc;
  };

  // Phase A: per key-column online max/sum over the query axis.
  for (int kti = 0; kti < NT / 8; ++kti) {
    int kt = wv + kti * 8;
    float m = NEG_INF, lsum = 0.f;
    for (int qt = kt; qt < NT; ++qt) {
      f32x8 s = score_tile(qt, kt);
      float tmax = s[0];
#pragma unroll
      for (int r = 1; r < 8; ++r) tmax = fmaxf(tmax, s[r]);
      tmax = fmaxf(tmax, __shfl_xor(tmax, 16, 32));   // combine lane pair (rows 0-7 | 8-15)
      float mn = fmaxf(m, tmax);
      float part = 0.f;
#pragma unroll
      for (int r = 0; r < 8; ++r) part += __expf(s[r] - mn);
      part += __shfl_xor(part, 16, 32);
      lsum = lsum * __expf(m - mn) + part;
      m = mn;
    }
    if (hi == 0) { sm[kt * 16 + n] = m; sl[kt * 16 + n] = 1.0f / lsum; }
  }
  __syncthreads();

  // Phase B: recompute P tiles, Z = P @ V (k in chunks of 32).
  for (int qti = 0; qti < NT / 8; ++qti) {
    int qt = wv + qti * 8;
    f32x8 zacc[4] = {};
    int nchunk = (qt + 2) >> 1;              // ceil((qt+1)/2) k-tiles of 32
    for (int c = 0; c < nchunk; ++c) {
#pragma unroll
      for (int half = 0; half < 2; ++half) {
        int kt = c * 2 + half;
        if (kt <= qt) {
          f32x8 s = score_tile(qt, kt);
          float mcol = sm[kt * 16 + n], lcol = sl[kt * 16 + n];
#pragma unroll
          for (int r = 0; r < 8; ++r)
            ptile[wv][(r + hi * 8) * 32 + half * 16 + n] =
                (__bf16)(__expf(s[r] - mcol) * lcol);
        } else {
#pragma unroll
          for (int r = 0; r < 8; ++r)
            ptile[wv][(r + hi * 8) * 32 + half * 16 + n] = (__bf16)0.f;
        }
      }
      asm volatile("s_wait_dscnt 0x0" ::: "memory");   // same-wave cross-lane LDS RAW
      bf16x16 a = load_frag_a(&ptile[wv][0], 32);
#pragma unroll
      for (int dt = 0; dt < 4; ++dt) {
        bf16x16 vb = load_frag_b(Vh + (size_t)dt * 16 * SEQ + c * 32, SEQ);
        zacc[dt] = wmma_bf16(a, vb, zacc[dt]);
      }
    }
#pragma unroll
    for (int dt = 0; dt < 4; ++dt)
#pragma unroll
      for (int r = 0; r < 8; ++r) {
        int q  = qt * 16 + r + hi * 8;
        int cc = h * DH + dt * 16 + n;
        Z[((size_t)b_ * SEQ + q) * DM + cc] = (__bf16)zacc[dt][r];
      }
  }
}

// ---- Output GEMM: out = Z @ Wo^T + bo (fp32 out), 32Mx64N per wave ----
__global__ void out_gemm_kernel(const __bf16* __restrict__ Zb,
                                const __bf16* __restrict__ Wo,
                                const float*  __restrict__ bo,
                                float* __restrict__ out) {
  const int NBLK = DM / 64;                  // 16 n-blocks of 64
  int t  = blockIdx.x * 8 + (threadIdx.x >> 5);
  int mb = t / NBLK, nb = t % NBLK;
  const __bf16* arow = Zb + (size_t)mb * 32 * DM;
  const __bf16* brow = Wo + (size_t)nb * 64 * DM;
  f32x8 acc[2][4] = {};
  for (int k = 0; k < DM; k += 32) {
    if (k + 64 < DM) {
      __builtin_prefetch(arow + k + 64, 0, 1);
      __builtin_prefetch(brow + k + 64, 0, 1);
    }
    bf16x16 a0 = load_frag_a(arow + k, DM);
    bf16x16 a1 = load_frag_a(arow + 16 * DM + k, DM);
#pragma unroll
    for (int j = 0; j < 4; ++j) {
      bf16x16 b = load_frag_b(brow + (size_t)j * 16 * DM + k, DM);
      acc[0][j] = wmma_bf16(a0, b, acc[0][j]);
      acc[1][j] = wmma_bf16(a1, b, acc[1][j]);
    }
  }
  int lane = threadIdx.x & 31;
  int hi = lane >> 4;
#pragma unroll
  for (int j = 0; j < 4; ++j) {
    int n = nb * 64 + j * 16 + (lane & 15);
    float bias = bo[n];
#pragma unroll
    for (int i = 0; i < 2; ++i)
#pragma unroll
      for (int r = 0; r < 8; ++r) {
        int m = mb * 32 + i * 16 + r + hi * 8;
        out[(size_t)m * DM + n] = acc[i][j][r] + bias;
      }
  }
}

extern "C" void kernel_launch(void* const* d_in, const int* in_sizes, int n_in,
                              void* d_out, int out_size, void* d_ws, size_t ws_size,
                              hipStream_t stream) {
  const float* x    = (const float*)d_in[0];
  const float* Wqkv = (const float*)d_in[1];
  const float* bqkv = (const float*)d_in[2];
  const float* Wo   = (const float*)d_in[3];
  const float* bo   = (const float*)d_in[4];
  float* out = (float*)d_out;

  char* ws = (char*)d_ws;
  const size_t MB = 1024 * 1024;
  __bf16* xn  = (__bf16*)(ws);                         // 16 MB  [8192,1024]
  __bf16* wqb = (__bf16*)(ws + 16 * MB);               //  6 MB  [3072,1024]
  __bf16* wob = (__bf16*)(ws + 22 * MB);               //  2 MB  [1024,1024]
  __bf16* Qb  = (__bf16*)(ws + 24 * MB);               // 16 MB  [b,h,s,d]
  __bf16* Kb  = (__bf16*)(ws + 40 * MB);               // 16 MB  [b,h,s,d]
  __bf16* Vtb = (__bf16*)(ws + 56 * MB);               // 16 MB  [b,h,d,s]
  __bf16* Zb  = (__bf16*)(ws + 72 * MB);               // 16 MB  [b,s,1024]

  rmsnorm_kernel<<<NB * SEQ, 256, 0, stream>>>(x, xn);
  f2bf_kernel<<<(NQKV * DM) / 256, 256, 0, stream>>>(Wqkv, wqb, NQKV * DM);
  f2bf_kernel<<<(DM * DM) / 256, 256, 0, stream>>>(Wo, wob, DM * DM);

  // QKV: 256 m-blocks(32) x 48 n-blocks(64), 8 waves/block, 1 block-tile/wave
  qkv_gemm_kernel<<<(256 * 48) / 8, 256, 0, stream>>>(xn, wqb, bqkv, Qb, Kb, Vtb);
  attention_kernel<<<NB * NH, 256, 0, stream>>>(Qb, Kb, Vtb, Zb);
  // OUT: 256 m-blocks(32) x 16 n-blocks(64)
  out_gemm_kernel<<<(256 * 16) / 8, 256, 0, stream>>>(Zb, wob, bo, out);
}
